// simpleEdgeModelAtt_82772609729150
// MI455X (gfx1250) — matrix operand order repaired and checked by reading, not compiled, not run on hardware
//
#include <hip/hip_runtime.h>

typedef __attribute__((ext_vector_type(16))) _Float16 v16h;
typedef __attribute__((ext_vector_type(8)))  _Float16 h8;
typedef __attribute__((ext_vector_type(4)))  _Float16 h4;
typedef __attribute__((ext_vector_type(8)))  float    v8f;

#define WMMA_F16(A, B, C) \
    __builtin_amdgcn_wmma_f32_16x16x32_f16(false, (A), false, (B), (short)0, (C), false, false)

// ---------------------------------------------------------------------------
// Pre-pack: convert f32 weight [K][N] into f16 WMMA B-fragment order.
// Fragment (kt,nt): lane L, element t holds w[kt*32 + (L>>4)*8 + t + (t&8)][nt*16 + (L&15)].
// Each thread packs one lane's 16 halves -> 32 contiguous bytes.
// ---------------------------------------------------------------------------
__global__ void pack_weights_kernel(const float* __restrict__ w,
                                    _Float16* __restrict__ dst, int K, int N)
{
    int idx = blockIdx.x * blockDim.x + threadIdx.x;
    int total = (K >> 5) * (N >> 4) * 32;
    if (idx >= total) return;
    int lane   = idx & 31;
    int tile   = idx >> 5;
    int ntiles = N >> 4;
    int nt = tile % ntiles;
    int kt = tile / ntiles;
    int m  = lane & 15;
    int hi = lane >> 4;
    const float* s = w + (size_t)(kt * 32 + hi * 8) * N + nt * 16 + m;
    _Float16* d = dst + (size_t)idx * 16;
#pragma unroll
    for (int t = 0; t < 16; ++t) d[t] = (_Float16)s[(size_t)(t + (t & 8)) * N];
}

// ---------------------------------------------------------------------------
// Main kernel: block = 256 threads (8 wave32), processes 64 edges.
// Wave w owns output N-tile w across 4 M-tiles of 16 edges.
// ---------------------------------------------------------------------------
__global__ __launch_bounds__(256)
void edge_mlp_att_kernel(const float* __restrict__ src,
                         const float* __restrict__ tgt,
                         const float* __restrict__ eat,
                         const v16h* __restrict__ w1p, const float* __restrict__ b1,
                         const v16h* __restrict__ w2p, const float* __restrict__ b2,
                         const v16h* __restrict__ wgp, const float* __restrict__ bg,
                         const v16h* __restrict__ wtp, const float* __restrict__ bt,
                         const v16h* __restrict__ wpp, const float* __restrict__ bp,
                         const float* __restrict__ wo, const float* __restrict__ bo,
                         float* __restrict__ out, int E)
{
    // 64KB static LDS, regions reused phase-to-phase.
    __shared__ __align__(16) unsigned char smem[49152 + 16384];
    _Float16* ldsX  = (_Float16*)smem;            // phase 1: concat input f16 [64][384]
    float*    ldsH2 = (float*)smem;               // phase 3+: h2 f32 [64][128]
    _Float16* ldsH1 = (_Float16*)(smem + 49152);  // phase 2: h1 f16 [64][128]
    float*    ldsG  = (float*)(smem + 49152);     // phase 4+: g  f32 [64][16]
    float*    ldsTH = ldsG + 64 * 16;             //           th f32 [64][16]
    float*    ldsPH = ldsTH + 64 * 16;            //           ph f32 [64][16]
    float*    ldsY  = ldsPH + 64 * 16;            //           y  f32 [64][16]

    const int base = blockIdx.x * 64;
    const int tid  = threadIdx.x;
    const int wave = tid >> 5;
    const int lane = tid & 31;
    const int m    = lane & 15;
    const int hi   = lane >> 4;

    // ---- Phase 1: cooperative load + concat + f32->f16 into LDS -------------
    for (int q = tid; q < 64 * 96; q += 256) {
        int row = q / 96;
        int col = (q - row * 96) * 4;
        int e = base + row; if (e >= E) e = E - 1;
        const float* p;
        if (col < 128)      p = src + (size_t)e * 128 + col;
        else if (col < 256) p = tgt + (size_t)e * 128 + (col - 128);
        else                p = eat + (size_t)e * 128 + (col - 256);
        float4 v = *(const float4*)p;
        h4 hv = { (_Float16)v.x, (_Float16)v.y, (_Float16)v.z, (_Float16)v.w };
        *(h4*)(ldsX + row * 384 + col) = hv;
    }
    __syncthreads();

    const int nt   = wave;
    const int ncol = nt * 16 + m;

    // ---- Phase 2: GEMM1  h1 = relu(X @ w1 + b1),  K=384 ---------------------
    v8f acc[4] = {};
    for (int k0 = 0; k0 < 384; k0 += 32) {
        v16h bf = w1p[((k0 >> 5) * 8 + nt) * 32 + lane];   // packed fragment
#pragma unroll
        for (int mt = 0; mt < 4; ++mt) {
            const _Float16* ar = ldsX + (mt * 16 + m) * 384 + k0 + hi * 8;
            h8 lo  = *(const h8*)ar;
            h8 hi8 = *(const h8*)(ar + 16);
            v16h af = __builtin_shufflevector(lo, hi8, 0,1,2,3,4,5,6,7,8,9,10,11,12,13,14,15);
            acc[mt] = WMMA_F16(af, bf, acc[mt]);
        }
    }
    {
        float bv = b1[ncol];
#pragma unroll
        for (int mt = 0; mt < 4; ++mt)
#pragma unroll
            for (int v = 0; v < 8; ++v) {
                float h = acc[mt][v] + bv;
                h = h > 0.f ? h : 0.f;
                ldsH1[(mt * 16 + v + 8 * hi) * 128 + ncol] = (_Float16)h;
            }
    }
    __syncthreads();

    // ---- Phase 3: GEMM2  h2 = h1 @ w2 + b2,  K=128 --------------------------
    v8f acc2[4] = {};
    for (int k0 = 0; k0 < 128; k0 += 32) {
        v16h bf = w2p[((k0 >> 5) * 8 + nt) * 32 + lane];
#pragma unroll
        for (int mt = 0; mt < 4; ++mt) {
            const _Float16* ar = ldsH1 + (mt * 16 + m) * 128 + k0 + hi * 8;
            h8 lo  = *(const h8*)ar;
            h8 hi8 = *(const h8*)(ar + 16);
            v16h af = __builtin_shufflevector(lo, hi8, 0,1,2,3,4,5,6,7,8,9,10,11,12,13,14,15);
            acc2[mt] = WMMA_F16(af, bf, acc2[mt]);
        }
    }
    {
        float bv = b2[ncol];
#pragma unroll
        for (int mt = 0; mt < 4; ++mt)
#pragma unroll
            for (int v = 0; v < 8; ++v)
                ldsH2[(mt * 16 + v + 8 * hi) * 128 + ncol] = acc2[mt][v] + bv;
    }
    __syncthreads();

    // ---- Phase 4: projections g/theta/phi = h2 @ W + b  (N=16, K=128) -------
#pragma unroll
    for (int rep = 0; rep < 2; ++rep) {
        int job = wave + rep * 8;                 // wave-uniform branch
        if (job < 12) {
            int which = job >> 2;                 // 0=g 1=theta 2=phi
            int mt    = job & 3;
            const v16h*  W  = (which == 0) ? wgp : (which == 1) ? wtp : wpp;
            const float* Bv = (which == 0) ? bg  : (which == 1) ? bt  : bp;
            float*       O  = (which == 0) ? ldsG : (which == 1) ? ldsTH : ldsPH;
            v8f a = {};
            for (int k0 = 0; k0 < 128; k0 += 32) {
                v16h bf = W[(k0 >> 5) * 32 + lane];
                const float* hp = ldsH2 + (mt * 16 + m) * 128 + k0 + hi * 8;
                float4 x0 = *(const float4*)(hp);
                float4 x1 = *(const float4*)(hp + 4);
                float4 x2 = *(const float4*)(hp + 16);
                float4 x3 = *(const float4*)(hp + 20);
                v16h af = { (_Float16)x0.x, (_Float16)x0.y, (_Float16)x0.z, (_Float16)x0.w,
                            (_Float16)x1.x, (_Float16)x1.y, (_Float16)x1.z, (_Float16)x1.w,
                            (_Float16)x2.x, (_Float16)x2.y, (_Float16)x2.z, (_Float16)x2.w,
                            (_Float16)x3.x, (_Float16)x3.y, (_Float16)x3.z, (_Float16)x3.w };
                a = WMMA_F16(af, bf, a);
            }
            float bb = Bv[m];
#pragma unroll
            for (int v = 0; v < 8; ++v)
                O[(mt * 16 + v + 8 * hi) * 16 + m] = a[v] + bb;
        }
    }
    __syncthreads();

    // ---- Phase 5: rank-1 attention + row softmax (VALU, f32) ----------------
#pragma unroll
    for (int r = 0; r < 4; ++r) {
        int R  = tid + 256 * r;
        int el = R >> 4;
        int i  = R & 15;
        float p = ldsPH[el * 16 + i];
        float tv[16], mx = -3.4e38f;
#pragma unroll
        for (int j = 0; j < 16; ++j) {
            tv[j] = p * ldsTH[el * 16 + j];
            mx = tv[j] > mx ? tv[j] : mx;
        }
        float s = 0.f, accy = 0.f;
#pragma unroll
        for (int j = 0; j < 16; ++j) {
            float ev = __expf(tv[j] - mx);
            s += ev;
            accy += ev * ldsG[el * 16 + j];
        }
        ldsY[el * 16 + i] = accy / s;
    }
    __syncthreads();

    // ---- Phase 6: out = y @ wo + bo + h2 (K=16, VALU, coalesced f32x4) ------
    for (int q = tid; q < 64 * 32; q += 256) {
        int row = q >> 5;
        int col = (q & 31) * 4;
        int e = base + row;
        if (e < E) {
            float a0 = bo[col], a1 = bo[col + 1], a2 = bo[col + 2], a3 = bo[col + 3];
            const float* yrow = ldsY + row * 16;
#pragma unroll
            for (int j = 0; j < 16; ++j) {
                float yj = yrow[j];
                float4 wv = *(const float4*)(wo + (size_t)j * 128 + col);
                a0 += yj * wv.x; a1 += yj * wv.y; a2 += yj * wv.z; a3 += yj * wv.w;
            }
            const float* hr = ldsH2 + row * 128 + col;
            a0 += hr[0]; a1 += hr[1]; a2 += hr[2]; a3 += hr[3];
            *(float4*)(out + (size_t)e * 128 + col) = make_float4(a0, a1, a2, a3);
        }
    }
}

extern "C" void kernel_launch(void* const* d_in, const int* in_sizes, int n_in,
                              void* d_out, int out_size, void* d_ws, size_t ws_size,
                              hipStream_t stream) {
    const float* src = (const float*)d_in[0];
    const float* tgt = (const float*)d_in[1];
    const float* eat = (const float*)d_in[2];
    const float* w1  = (const float*)d_in[3];
    const float* b1  = (const float*)d_in[4];
    const float* w2  = (const float*)d_in[5];
    const float* b2  = (const float*)d_in[6];
    const float* wg  = (const float*)d_in[7];
    const float* bg  = (const float*)d_in[8];
    const float* wt  = (const float*)d_in[9];
    const float* bt  = (const float*)d_in[10];
    const float* wp  = (const float*)d_in[11];
    const float* bp  = (const float*)d_in[12];
    const float* wo  = (const float*)d_in[13];
    const float* bo  = (const float*)d_in[14];
    float* out = (float*)d_out;

    // Workspace layout (f16 packed weights, fragment order):
    //   w1p: 384x128 -> 98304 B   @ 0
    //   w2p: 128x128 -> 32768 B   @ 98304
    //   wgp/wtp/wpp: 128x16 -> 4096 B each @ 131072 / 135168 / 139264
    _Float16* w1p = (_Float16*)d_ws;
    _Float16* w2p = (_Float16*)((char*)d_ws + 98304);
    _Float16* wgp = (_Float16*)((char*)d_ws + 131072);
    _Float16* wtp = (_Float16*)((char*)d_ws + 135168);
    _Float16* wpp = (_Float16*)((char*)d_ws + 139264);

    pack_weights_kernel<<<(12 * 8 * 32 + 255) / 256, 256, 0, stream>>>(w1, w1p, 384, 128);
    pack_weights_kernel<<<(4 * 8 * 32 + 255) / 256, 256, 0, stream>>>(w2, w2p, 128, 128);
    pack_weights_kernel<<<1, 128, 0, stream>>>(wg, wgp, 128, 16);
    pack_weights_kernel<<<1, 128, 0, stream>>>(wt, wtp, 128, 16);
    pack_weights_kernel<<<1, 128, 0, stream>>>(wp, wpp, 128, 16);

    const int E = in_sizes[0] / 128;
    const int blocks = (E + 63) / 64;
    edge_mlp_att_kernel<<<blocks, 256, 0, stream>>>(
        src, tgt, eat,
        (const v16h*)w1p, b1, (const v16h*)w2p, b2,
        (const v16h*)wgp, bg, (const v16h*)wtp, bt, (const v16h*)wpp, bp,
        wo, bo, out, E);
}
